// TimeAware_LVCBlock_11381663335079
// MI455X (gfx1250) — matrix operand order repaired and checked by reading, not compiled, not run on hardware
//
#include <hip/hip_runtime.h>
#include <hip/hip_bf16.h>
#include <math.h>

typedef __attribute__((ext_vector_type(16))) __bf16 v16bf;
typedef __attribute__((ext_vector_type(8)))  float  v8f;

#define B_      16
#define T_      8192
#define L_      32
#define HOP_    256
#define CIN_    64
#define COUT_   128
#define KS_     3
#define NLAYER_ 4
#define CCH_    80
#define HID_    64
#define EMB_    512
#define LW_     (CIN_*COUT_*KS_*NLAYER_)   // 98304
#define KDIM_   (HID_*KS_)                 // 192
#define NTOT_   (B_*L_)                    // 512
#define MSLICE_ (CIN_*COUT_*KS_)           // 24576

// ---------------- WMMA helpers (gfx1250, wave32) ----------------

__device__ __forceinline__ v8f wmma_bf16(v16bf a, v16bf b, v8f c) {
  // D(f32 16x16) = A(16x32 bf16) x B(32x16 bf16) + C
  return __builtin_amdgcn_wmma_f32_16x16x32_bf16(false, a, false, b, (short)0, c,
                                                 false, false);
}

// A fragment: lane holds row M = lane&15. elems 0..15 <-> K[c0..c0+15],
// c0 = 16*(lane>=16).   p = row base in LDS.
__device__ __forceinline__ v16bf load_a(const __bf16* __restrict__ p, int kbase, int hi) {
  v16bf a;
  const int c0 = kbase + (hi ? 16 : 0);
#pragma unroll
  for (int j = 0; j < 16; ++j) a[j] = p[c0 + j];
  return a;
}

// B fragment: lane holds col N = lane&15; lanes 0-15: K[kbase..kbase+15],
// lanes 16-31: K[kbase+16..kbase+31].  p = column base (K-contiguous) in LDS.
__device__ __forceinline__ v16bf load_b(const __bf16* __restrict__ p, int kbase, int hi) {
  v16bf bv;
  const int c0 = kbase + (hi ? 16 : 0);
#pragma unroll
  for (int j = 0; j < 16; ++j) bv[j] = p[c0 + j];
  return bv;
}

// low 32 bits of a generic pointer to __shared__ == raw LDS byte offset
__device__ __forceinline__ unsigned lds_off_of(const void* p) {
  return (unsigned)(unsigned long long)p;
}

// ---------------- Kernel 1: predictor trunk (tiny) ----------------

__global__ __launch_bounds__(256)
void kp_predictor(const float* __restrict__ c,   const float* __restrict__ dstep,
                  const float* __restrict__ fcw, const float* __restrict__ fcb,
                  const float* __restrict__ inw, const float* __restrict__ inb,
                  const float* __restrict__ resw,const float* __restrict__ resb,
                  const float* __restrict__ kbw, const float* __restrict__ kbb,
                  float* __restrict__ hcol, float* __restrict__ biasout)
{
  __shared__ float noise[CCH_];
  __shared__ float cond[CCH_ * L_];
  __shared__ float hin[HID_ * L_];
  __shared__ float ra[HID_ * L_];
  __shared__ float rb[HID_ * L_];
  const int b = blockIdx.x, tid = threadIdx.x;

  if (tid < CCH_) {
    float acc = fcb[tid];
    const float* dp = dstep + b * EMB_;
    const float* wp = fcw + tid * EMB_;
    for (int e = 0; e < EMB_; ++e) acc += dp[e] * wp[e];
    noise[tid] = acc;
  }
  __syncthreads();

  for (int idx = tid; idx < CCH_ * L_; idx += 256) {
    int ch = idx / L_, l = idx % L_;
    cond[idx] = c[(b * CCH_ + ch) * L_ + l] + noise[ch];
  }
  __syncthreads();

  // kp_in conv: (64,80,5), pad 2, lrelu
  for (int idx = tid; idx < HID_ * L_; idx += 256) {
    int o = idx / L_, t = idx % L_;
    float acc = inb[o];
    for (int i = 0; i < CCH_; ++i)
      for (int k = 0; k < 5; ++k) {
        int p = t + k - 2;
        if (p >= 0 && p < L_) acc += cond[i * L_ + p] * inw[(o * CCH_ + i) * 5 + k];
      }
    hin[idx] = acc > 0.f ? acc : 0.1f * acc;
  }
  __syncthreads();

  // 6 residual convs (64,64,3), pad 1, lrelu
  const float* rsrc = hin;
  float* rdst = ra;
  for (int j = 0; j < 6; ++j) {
    for (int idx = tid; idx < HID_ * L_; idx += 256) {
      int o = idx / L_, t = idx % L_;
      float acc = resb[j * HID_ + o];
      const float* wr = resw + ((j * HID_ + o) * HID_) * KS_;
      for (int i = 0; i < HID_; ++i)
        for (int k = 0; k < KS_; ++k) {
          int p = t + k - 1;
          if (p >= 0 && p < L_) acc += rsrc[i * L_ + p] * wr[i * KS_ + k];
        }
      rdst[idx] = acc > 0.f ? acc : 0.1f * acc;
    }
    __syncthreads();
    rsrc = rdst;
    rdst = (rdst == ra) ? rb : ra;
  }
  float* h = rdst;  // free buffer; rsrc holds final r
  for (int idx = tid; idx < HID_ * L_; idx += 256) h[idx] = hin[idx] + rsrc[idx];
  __syncthreads();

  // Hcol[(i*3+k)*512 + b*32 + l] = h[i][l+k-1] (zero pad)
  for (int idx = tid; idx < HID_ * KS_ * L_; idx += 256) {
    int i = idx / (KS_ * L_), rem = idx % (KS_ * L_);
    int k = rem / L_, l = rem % L_;
    int p = l + k - 1;
    float v = (p >= 0 && p < L_) ? h[i * L_ + p] : 0.f;
    hcol[(i * KS_ + k) * NTOT_ + b * L_ + l] = v;
  }

  // bias conv: (512,64,3) pad 1 -> biasout[b][layer][oc][l]
  for (int idx = tid; idx < 512 * L_; idx += 256) {
    int o = idx / L_, t = idx % L_;
    float acc = kbb[o];
    const float* wb = kbw + o * HID_ * KS_;
    for (int i = 0; i < HID_; ++i)
      for (int k = 0; k < KS_; ++k) {
        int p = t + k - 1;
        if (p >= 0 && p < L_) acc += h[i * L_ + p] * wb[i * KS_ + k];
      }
    int layer = o >> 7, oc = o & 127;
    biasout[((b * NLAYER_ + layer) * COUT_ + oc) * L_ + t] = acc;
  }
}

// ---------------- Kernel 2: kp_k conv as WMMA GEMM ----------------
// Out[n=512][r'=98304](bf16) = Wperm(98304x192) @ Hcol(192x512) + kp_k_b
// Rows permuted to r' = layer*24576 + o*192 + cin*3 + ktap so each LVC slice is
// a contiguous bf16 blob in exactly the LDS layout the LVC kernel consumes.

#define AS_   200   // LDS row stride (bf16 elems): 400B rows, bank-partitioning
#define NBLK_ 128

__global__ __launch_bounds__(256)
void gemm_kpred(const float* __restrict__ kkw, const float* __restrict__ kkb,
                const float* __restrict__ hcol, __bf16* __restrict__ kern)
{
  __shared__ __bf16 ldsA[128 * AS_];   // permuted W rows [rlocal][k]
  __shared__ __bf16 ldsB[NBLK_ * AS_]; // Hcol^T          [nlocal][k]
  __shared__ float  ldsD[16 * 132];    // transpose staging
  const int tid = threadIdx.x;
  const int mbase = blockIdx.x * 128;  // base of permuted row index r'
  const int nbase = blockIdx.y * NBLK_;

  // stage A with row permutation r' -> m (row payloads stay 192-contiguous)
  for (int i = 0; i < (128 * KDIM_) / 256; ++i) {
    int e = tid + i * 256;
    int rl = e / KDIM_, kh = e % KDIM_;
    int rp = mbase + rl;
    int layer = rp / MSLICE_, rem = rp % MSLICE_;
    int o = rem / KDIM_, ck = rem % KDIM_;
    int cin = ck / KS_, kt = ck % KS_;
    int m = ((layer * CIN_ + cin) * COUT_ + o) * KS_ + kt;
    ldsA[rl * AS_ + kh] = (__bf16)kkw[m * KDIM_ + kh];
  }
  // stage B (Hcol is [k][n], n contiguous -> coalesced)
  for (int i = 0; i < (KDIM_ * NBLK_) / 256; ++i) {
    int e = tid + i * 256;
    int k = e / NBLK_, n = e % NBLK_;
    ldsB[n * AS_ + k] = (__bf16)hcol[k * NTOT_ + nbase + n];
  }
  __syncthreads();

  const int wave = tid >> 5, lane = tid & 31, ln = lane & 15, hi = lane >> 4;
  float biasr[8];
#pragma unroll
  for (int r = 0; r < 8; ++r) {
    int rp = mbase + wave * 16 + (hi ? 8 : 0) + r;
    int layer = rp / MSLICE_, rem = rp % MSLICE_;
    int o = rem / KDIM_, ck = rem % KDIM_;
    int cin = ck / KS_, kt = ck % KS_;
    biasr[r] = kkb[((layer * CIN_ + cin) * COUT_ + o) * KS_ + kt];
  }

  // hoist A fragments (reused by all 8 N-subtiles)
  const __bf16* ap = &ldsA[(wave * 16 + ln) * AS_];
  v16bf afr[KDIM_ / 32];
#pragma unroll
  for (int s = 0; s < KDIM_ / 32; ++s) afr[s] = load_a(ap, s * 32, hi);

  for (int nt = 0; nt < NBLK_ / 16; ++nt) {
    v8f acc;
#pragma unroll
    for (int r = 0; r < 8; ++r) acc[r] = biasr[r];
    const __bf16* bp = &ldsB[(nt * 16 + ln) * AS_];
#pragma unroll
    for (int s = 0; s < KDIM_ / 32; ++s) {
      v16bf bb = load_b(bp, s * 32, hi);
      acc = wmma_bf16(afr[s], bb, acc);
    }
    // transpose through LDS -> coalesced [n][r'] bf16 global stores
#pragma unroll
    for (int r = 0; r < 8; ++r)
      ldsD[ln * 132 + wave * 16 + (hi ? 8 : 0) + r] = acc[r];
    __syncthreads();
    {
      int ns = tid >> 4, mo = (tid & 15) * 8;
      int ng = nbase + nt * 16 + ns;
      __bf16* op = kern + (size_t)ng * LW_ + mbase + mo;
#pragma unroll
      for (int j = 0; j < 8; ++j) op[j] = (__bf16)ldsD[ns * 132 + mo + j];
    }
    __syncthreads();
  }
}

// ---------------- Kernel 3: LVC layer (async-LDS + WMMA + fused gate) ----------------

#define WSP_ 200  // ldsW row stride (bf16): 400B rows, 16B aligned, bank-partitioning
#define XS_  288  // ldsX row stride (bf16), win <= 272

__global__ __launch_bounds__(256)
void lvc_layer(const float* __restrict__ xin, float* __restrict__ xout,
               const __bf16* __restrict__ kern, const float* __restrict__ biasws,
               int layer, int dil, float scale)
{
  __shared__ __bf16 ldsW[COUT_ * WSP_]; // [o][cin*3+k] (rows padded to 400B)
  __shared__ __bf16 ldsX[CIN_ * XS_];   // [cin][w]
  const int l = blockIdx.x, b = blockIdx.y, tid = threadIdx.x;

  // --- async DMA of the predicted-kernel slice: 3072 x 16B chunks, no transpose,
  // --- no conversion (slice already bf16 in [o][cin*3+k] row order).
  {
    const __bf16* wslice = kern + (size_t)(b * L_ + l) * LW_ + layer * MSLICE_;
    const int CPR = (KDIM_ * 2) / 16;  // 24 chunks per 384B row
#pragma unroll
    for (int j = 0; j < (MSLICE_ * 2) / 16 / 256; ++j) {  // 12 chunks/thread
      int cchunk = tid + j * 256;
      int row = cchunk / CPR, off = cchunk % CPR;
      unsigned loff = lds_off_of(&ldsW[row * WSP_ + off * 8]);
      const __bf16* g = wslice + cchunk * 8;
      asm volatile("global_load_async_to_lds_b128 %0, %1, off"
                   :: "v"(loff), "v"(g) : "memory");
    }
  }

  // stage x window with halo dil each side (zero-padded at sequence ends)
  const int win = HOP_ + 2 * dil;
  for (int e = tid; e < CIN_ * win; e += 256) {
    int cin = e / win, w = e % win;
    int t = l * HOP_ - dil + w;
    float v = (t >= 0 && t < T_) ? xin[(b * CIN_ + cin) * T_ + t] : 0.f;
    ldsX[cin * XS_ + w] = (__bf16)v;
  }

  asm volatile("s_wait_asynccnt 0" ::: "memory");
  __syncthreads();

  const int wave = tid >> 5, lane = tid & 31, ln = lane & 15, hi = lane >> 4;
  const int gb = wave & 3;            // gate channel block (0..3); filt = +64
  const int nt0 = (wave >> 2) * 8;    // halves of the 16 position tiles

  float bg[8], bf_[8];
  const float* bp = biasws + ((b * NLAYER_ + layer) * COUT_) * L_;
#pragma unroll
  for (int r = 0; r < 8; ++r) {
    int mg = gb * 16 + (hi ? 8 : 0) + r;
    bg[r]  = bp[mg * L_ + l];
    bf_[r] = bp[(mg + 64) * L_ + l];
  }

  // hoist gate+filt A fragments (reused by all 8 position tiles)
  const __bf16* apg = &ldsW[(gb * 16 + ln) * WSP_];
  const __bf16* apf = &ldsW[((64 + gb * 16) + ln) * WSP_];
  v16bf ag6[KDIM_ / 32], af6[KDIM_ / 32];
#pragma unroll
  for (int s = 0; s < KDIM_ / 32; ++s) {
    ag6[s] = load_a(apg, s * 32, hi);
    af6[s] = load_a(apf, s * 32, hi);
  }

  for (int ni = 0; ni < 8; ++ni) {
    const int nt = nt0 + ni;
    const int pos = nt * 16 + ln;     // lane's output position (B-fragment col)
    v8f accg, accf;
#pragma unroll
    for (int r = 0; r < 8; ++r) { accg[r] = bg[r]; accf[r] = bf_[r]; }
#pragma unroll
    for (int s = 0; s < KDIM_ / 32; ++s) {
      v16bf xb;
#pragma unroll
      for (int j = 0; j < 16; ++j) {
        int kidx = s * 32 + (hi ? 16 : 0) + j;  // cin*3 + k
        int cin = kidx / KS_, kq = kidx % KS_;
        xb[j] = ldsX[cin * XS_ + pos + kq * dil];
      }
      accg = wmma_bf16(ag6[s], xb, accg);
      accf = wmma_bf16(af6[s], xb, accf);
    }
    // fused residual gate in registers
#pragma unroll
    for (int r = 0; r < 8; ++r) {
      int mg = gb * 16 + (hi ? 8 : 0) + r;
      int t = l * HOP_ + pos;
      float xold = xin[(b * CIN_ + mg) * T_ + t];
      float s_ = 1.0f / (1.0f + __expf(-accg[r]));
      xout[(b * CIN_ + mg) * T_ + t] = (xold + s_ * tanhf(accf[r])) * scale;
    }
  }
}

// ---------------- launch ----------------

extern "C" void kernel_launch(void* const* d_in, const int* in_sizes, int n_in,
                              void* d_out, int out_size, void* d_ws, size_t ws_size,
                              hipStream_t stream) {
  const float* x   = (const float*)d_in[0];
  const float* c   = (const float*)d_in[1];
  const float* ds  = (const float*)d_in[2];
  // d_in[3] = global_cond: unused by the reference
  const float* fcw = (const float*)d_in[4];
  const float* fcb = (const float*)d_in[5];
  const float* inw = (const float*)d_in[6];
  const float* inb = (const float*)d_in[7];
  const float* rsw = (const float*)d_in[8];
  const float* rsb = (const float*)d_in[9];
  const float* kkw = (const float*)d_in[10];
  const float* kkb = (const float*)d_in[11];
  const float* kbw = (const float*)d_in[12];
  const float* kbb = (const float*)d_in[13];
  float* out = (float*)d_out;

  float* ws     = (float*)d_ws;
  float* hcol   = ws;                                    // 192*512 f32
  float* biasws = hcol + KDIM_ * NTOT_;                  // 16*4*128*32 f32
  __bf16* kern  = (__bf16*)(biasws + B_ * NLAYER_ * COUT_ * L_); // 512*98304 bf16
  float* x0     = (float*)(kern + (size_t)NTOT_ * LW_);  // ping
  float* x1     = x0 + (size_t)B_ * CIN_ * T_;           // pong

  kp_predictor<<<B_, 256, 0, stream>>>(c, ds, fcw, fcb, inw, inb, rsw, rsb,
                                       kbw, kbb, hcol, biasws);
  gemm_kpred<<<dim3(LW_ / 128, NTOT_ / NBLK_), 256, 0, stream>>>(kkw, kkb, hcol, kern);
  lvc_layer<<<dim3(L_, B_), 256, 0, stream>>>(x,  x0,  kern, biasws, 0, 1, 1.0f);
  lvc_layer<<<dim3(L_, B_), 256, 0, stream>>>(x0, x1,  kern, biasws, 1, 2, 1.0f);
  lvc_layer<<<dim3(L_, B_), 256, 0, stream>>>(x1, x0,  kern, biasws, 2, 4, 1.0f);
  lvc_layer<<<dim3(L_, B_), 256, 0, stream>>>(x0, out, kern, biasws, 3, 8, 0.70710678118f);
}
